// Attention_13022340841897
// MI455X (gfx1250) — matrix operand order repaired and checked by reading, not compiled, not run on hardware
//
#include <hip/hip_runtime.h>

// ---------------- problem constants ----------------
// B=2, M=8, I=128 -> NQ=2048 query rows; J=64 memories/query; DIM=512;
// HEADS=8, DIM_HEAD=64, INNER=512; SCALE = 1/8.
#define NQ      2048
#define JDIM    64
#define DMODEL  512
#define HEADS   8
#define PW      4096          // HEADS * DMODEL (concat-over-heads width)
#define SCALE_F 0.125f

typedef __attribute__((ext_vector_type(16))) __bf16 bf16x16;
typedef __attribute__((ext_vector_type(8)))  float  v8f;
typedef __attribute__((ext_vector_type(4)))  float  f32x4;

struct alignas(16) U128 { unsigned int x, y, z, w; };
struct alignas(8)  U64  { unsigned int x, y; };
struct alignas(8)  US4  { unsigned short a, b, c, d; };

__device__ __forceinline__ unsigned short f2bf(float f) {
  unsigned int u = __float_as_uint(f);
  unsigned int r = u + 0x7FFFu + ((u >> 16) & 1u);   // round-to-nearest-even
  return (unsigned short)(r >> 16);
}

__device__ __forceinline__ bf16x16 bfpair(U128 a, U128 b) {
  union { U128 q[2]; bf16x16 v; } u;
  u.q[0] = a; u.q[1] = b;
  return u.v;
}

// ---------------- f32 -> bf16 bulk convert (x) ----------------
__global__ __launch_bounds__(256)
void f32_to_bf16_vec4(const float* __restrict__ in, unsigned short* __restrict__ out, int n4) {
  int i = blockIdx.x * 256 + threadIdx.x;
  if (i < n4) {
    f32x4 f = ((const f32x4*)in)[i];
    US4 o;
    o.a = f2bf(f.x); o.b = f2bf(f.y); o.c = f2bf(f.z); o.d = f2bf(f.w);
    ((US4*)out)[i] = o;
  }
}

// ---------------- prep: Astack[d', h*512+d] = SCALE * sum_t Wq[d',h64+t]*Wkv[d,h64+t] ----------------
__global__ __launch_bounds__(256)
void prep_A(const float* __restrict__ Wq, const float* __restrict__ Wkv,
            unsigned short* __restrict__ Abf) {
  __shared__ float q_s[64][65];
  __shared__ float k_s[64][65];
  int bid = blockIdx.x;                 // 512 blocks: h(8) x dt(8) x et(8)
  int h  = bid >> 6;
  int dt = (bid >> 3) & 7;
  int et = bid & 7;
  int d0 = dt * 64, e0 = et * 64;
  int t = threadIdx.x;
  for (int idx = t; idx < 4096; idx += 256) {
    int r = idx >> 6, c = idx & 63;
    q_s[r][c] = Wq [(d0 + r) * 512  + h * 64 + c];
    k_s[r][c] = Wkv[(e0 + r) * 1024 + h * 64 + c];
  }
  __syncthreads();
  int dl  = t & 63;
  int el0 = (t >> 6) * 16;
  for (int kk = 0; kk < 16; kk++) {
    int el = el0 + kk;
    float s = 0.f;
#pragma unroll 8
    for (int c = 0; c < 64; c++) s = fmaf(q_s[dl][c], k_s[el][c], s);
    Abf[(size_t)(d0 + dl) * PW + h * 512 + e0 + el] = f2bf(s * SCALE_F);
  }
}

// ---------------- prep: W2[h*512+d, e] = sum_t Wkv[d, 512+h64+t] * Wo[h64+t, e] ----------------
__global__ __launch_bounds__(256)
void prep_W2(const float* __restrict__ Wkv, const float* __restrict__ Wo,
             unsigned short* __restrict__ W2bf) {
  __shared__ float v_s[64][65];   // [d_local][t]
  __shared__ float o_s[64][65];   // [t][e_local]
  int bid = blockIdx.x;
  int h  = bid >> 6;
  int dt = (bid >> 3) & 7;
  int et = bid & 7;
  int d0 = dt * 64, e0 = et * 64;
  int t = threadIdx.x;
  for (int idx = t; idx < 4096; idx += 256) {
    int r = idx >> 6, c = idx & 63;
    v_s[r][c] = Wkv[(d0 + r) * 1024 + 512 + h * 64 + c];
    o_s[r][c] = Wo [(h * 64 + r) * 512 + e0 + c];
  }
  __syncthreads();
  int dl  = t & 63;
  int el0 = (t >> 6) * 16;
  for (int kk = 0; kk < 16; kk++) {
    int el = el0 + kk;
    float s = 0.f;
#pragma unroll 8
    for (int c = 0; c < 64; c++) s = fmaf(v_s[dl][c], o_s[c][el], s);
    W2bf[(size_t)(h * 512 + d0 + dl) * 512 + e0 + el] = f2bf(s);
  }
}

// ---------------- generic bf16 WMMA GEMM: C[M,N] = A[M,K] @ B[K,N] (+bias) ----------------
// block tile 128x128, K-chunk 32, 8 waves (wave w owns M-subtile w, loops 8 N-subtiles)
// A tile staged with GLOBAL_LOAD_ASYNC_TO_LDS_B128 (ASYNCcnt), B tile transposed in regs.
template <bool BF16_OUT>
__global__ __launch_bounds__(256)
void gemm_bf16(const unsigned short* __restrict__ A, const unsigned short* __restrict__ B,
               void* __restrict__ Cptr, const float* __restrict__ bias,
               int Mdim, int Ndim, int Kdim) {
  __shared__ __align__(16) unsigned short As[128 * 40];  // row pitch 40 u16 = 80B (32 used)
  __shared__ __align__(16) unsigned short Bt[128 * 40];  // B transposed: [n_local][k_local]
  int m0 = blockIdx.y * 128, n0 = blockIdx.x * 128;
  int t = threadIdx.x;
  int wave = t >> 5, lane = t & 31, lm = lane & 15, hi = lane >> 4;

  v8f acc[8];
#pragma unroll
  for (int i = 0; i < 8; i++) acc[i] = (v8f){0.f,0.f,0.f,0.f,0.f,0.f,0.f,0.f};

  // thread -> (row-pair, col-chunk) map for the B transpose stage:
  // lanes advance rp (4B LDS stride) => conflict-free b32 stores
  int b_c8 = t >> 4;      // 0..15
  int b_rp = t & 15;      // 0..15 (row pair)

  for (int k0 = 0; k0 < Kdim; k0 += 32) {
    __syncthreads();
    // stage A tile [128 x 32] via async copy global->LDS (16B per lane per op)
#pragma unroll
    for (int idx = t; idx < 512; idx += 256) {
      int r = idx >> 2, c8 = idx & 3;
      unsigned int lds_off = (unsigned int)(size_t)(&As[r * 40 + c8 * 8]);
      unsigned long long ga =
          (unsigned long long)(size_t)(A + (size_t)(m0 + r) * Kdim + k0 + c8 * 8);
      asm volatile("global_load_async_to_lds_b128 %0, %1, off"
                   :: "v"(lds_off), "v"(ga) : "memory");
    }
    // stage B tile [32 x 128] transposed into Bt[n][k]; pack row pairs as b32
    {
      U128 v0 = *(const U128*)(B + (size_t)(k0 + 2 * b_rp    ) * Ndim + n0 + b_c8 * 8);
      U128 v1 = *(const U128*)(B + (size_t)(k0 + 2 * b_rp + 1) * Ndim + n0 + b_c8 * 8);
      const unsigned short* s0 = (const unsigned short*)&v0;
      const unsigned short* s1 = (const unsigned short*)&v1;
#pragma unroll
      for (int z = 0; z < 8; z++) {
        unsigned int pack = (unsigned int)s0[z] | ((unsigned int)s1[z] << 16);
        *(unsigned int*)(&Bt[(b_c8 * 8 + z) * 40 + 2 * b_rp]) = pack;
      }
    }
    asm volatile("s_wait_asynccnt 0" ::: "memory");   // LDS writes from async copy done
    __syncthreads();
    // A fragment for this wave's M-subtile (lanes<16: K 0-7,16-23; lanes>=16: K 8-15,24-31)
    const unsigned short* ar = &As[(wave * 16 + lm) * 40 + (hi ? 8 : 0)];
    bf16x16 afrag = bfpair(*(const U128*)ar, *(const U128*)(ar + 16));
#pragma unroll
    for (int ns = 0; ns < 8; ns++) {
      const unsigned short* br = &Bt[(ns * 16 + lm) * 40 + (hi ? 8 : 0)];
      bf16x16 bfrag = bfpair(*(const U128*)br, *(const U128*)(br + 16));
      acc[ns] = __builtin_amdgcn_wmma_f32_16x16x32_bf16(
          false, afrag, false, bfrag, (short)0, acc[ns], false, false);
    }
  }
  // writeback: D reg r -> row M = r + (hi?8:0), col = lm within 16x16 tile
#pragma unroll
  for (int ns = 0; ns < 8; ns++) {
#pragma unroll
    for (int r = 0; r < 8; r++) {
      int row = m0 + wave * 16 + r + (hi ? 8 : 0);
      int col = n0 + ns * 16 + lm;
      float v = acc[ns][r];
      if (BF16_OUT) {
        ((unsigned short*)Cptr)[(size_t)row * Ndim + col] = f2bf(v);
      } else {
        ((float*)Cptr)[(size_t)row * Ndim + col] = v + bias[col];
      }
    }
  }
}

// ---------------- fused attention: one block per query row n = (b*M + m)*I + i ----------------
// sim[h,j] = P[n,h,:] . mems[n,j,:]; masked softmax over j; c_h = attn_h @ mems -> Cbf[n, h*512+e]
#define MP  520  // memsb row pitch (u16): [j][d], 16B-aligned rows, bank-spreading
#define MTP 68   // memsT row pitch (u16): [d][j], 8B-aligned rows, gcd(34,64)=2 -> spread
__global__ __launch_bounds__(256)
void attn_kernel(const float* __restrict__ mems, const int* __restrict__ mask,
                 const unsigned short* __restrict__ Pbf, unsigned short* __restrict__ Cbf) {
  __shared__ __align__(16) unsigned short memsb[64 * MP];    // 65 KB bf16 mems [j][d]
  __shared__ __align__(16) unsigned short memsT[512 * MTP];  // 68 KB bf16 mems [d][j]
  __shared__ __align__(16) unsigned short Ps[16 * 512];      // 16 KB P rows, rows 8..15 zero
  __shared__ __align__(16) float          simP[2][8 * 64];   //  4 KB K-split partial scores
  __shared__ __align__(16) unsigned short attnS[16 * 64];    //  2 KB bf16 attn, rows 8..15 zero
  int n = blockIdx.x;
  int t = threadIdx.x;
  int wave = t >> 5, lane = t & 31, lm = lane & 15, hi = lane >> 4;

  U128 z128; z128.x = 0; z128.y = 0; z128.z = 0; z128.w = 0;

  // stage P[n] : rows 0..7 real, rows 8..15 zero (kills divergent A-frag padding)
  const unsigned short* Pn = Pbf + (size_t)n * PW;
  for (int idx = t; idx < 1024; idx += 256) {
    int h = idx >> 6, c8 = idx & 63;
    U128 v = (h < 8) ? *(const U128*)(Pn + h * 512 + c8 * 8) : z128;
    *(U128*)(&Ps[h * 512 + c8 * 8]) = v;
  }
  // zero attnS rows 8..15 once (softmax fills rows 0..7 later)
  if (t < 64) *(U128*)(&attnS[512 + t * 8]) = z128;

  // stage mems[n]: 64x512 f32 -> bf16, row-major AND transposed tiles.
  // thread handles a 4x4 (j x d) micro-tile; non-temporal loads (256MB stream > L2).
  const float* Mn = mems + (size_t)n * (JDIM * DMODEL);
#pragma unroll
  for (int it = 0; it < 8; it++) {
    int idx = t + it * 256;          // 0..2047
    int jq = idx & 15;               // j-quad
    int c4 = idx >> 4;               // 0..127 d-quad
    int j0 = jq * 4, d0 = c4 * 4;
    const float* src = Mn + (size_t)j0 * DMODEL + d0;
    f32x4 f0 = __builtin_nontemporal_load((const f32x4*)(src));
    f32x4 f1 = __builtin_nontemporal_load((const f32x4*)(src +     DMODEL));
    f32x4 f2 = __builtin_nontemporal_load((const f32x4*)(src + 2 * DMODEL));
    f32x4 f3 = __builtin_nontemporal_load((const f32x4*)(src + 3 * DMODEL));
    unsigned short b00=f2bf(f0.x), b01=f2bf(f0.y), b02=f2bf(f0.z), b03=f2bf(f0.w);
    unsigned short b10=f2bf(f1.x), b11=f2bf(f1.y), b12=f2bf(f1.z), b13=f2bf(f1.w);
    unsigned short b20=f2bf(f2.x), b21=f2bf(f2.y), b22=f2bf(f2.z), b23=f2bf(f2.w);
    unsigned short b30=f2bf(f3.x), b31=f2bf(f3.y), b32=f2bf(f3.z), b33=f2bf(f3.w);
    US4 r0{b00,b01,b02,b03}; *(US4*)(&memsb[(j0+0)*MP + d0]) = r0;
    US4 r1{b10,b11,b12,b13}; *(US4*)(&memsb[(j0+1)*MP + d0]) = r1;
    US4 r2{b20,b21,b22,b23}; *(US4*)(&memsb[(j0+2)*MP + d0]) = r2;
    US4 r3{b30,b31,b32,b33}; *(US4*)(&memsb[(j0+3)*MP + d0]) = r3;
    US4 c0{b00,b10,b20,b30}; *(US4*)(&memsT[(d0+0)*MTP + j0]) = c0;
    US4 c1{b01,b11,b21,b31}; *(US4*)(&memsT[(d0+1)*MTP + j0]) = c1;
    US4 c2{b02,b12,b22,b32}; *(US4*)(&memsT[(d0+2)*MTP + j0]) = c2;
    US4 c3{b03,b13,b23,b33}; *(US4*)(&memsT[(d0+3)*MTP + j0]) = c3;
  }
  __syncthreads();

  // ---- sim GEMM: [16(8 used) x 64] = P[16 x 512] x memsT ----
  // all 8 waves: wave = (khalf<<2) | jtile ; each does 8 K-steps of 32
  {
    int jt    = wave & 3;
    int khalf = wave >> 2;
    v8f acc = (v8f){0.f,0.f,0.f,0.f,0.f,0.f,0.f,0.f};
#pragma unroll
    for (int kc = 0; kc < 8; kc++) {
      int k0 = khalf * 256 + kc * 32;
      const unsigned short* pr = &Ps[lm * 512 + k0 + (hi ? 8 : 0)];
      bf16x16 af = bfpair(*(const U128*)pr, *(const U128*)(pr + 16));
      const unsigned short* br = &memsb[(jt * 16 + lm) * MP + k0 + (hi ? 8 : 0)];
      bf16x16 bv = bfpair(*(const U128*)br, *(const U128*)(br + 16));
      acc = __builtin_amdgcn_wmma_f32_16x16x32_bf16(
          false, af, false, bv, (short)0, acc, false, false);
    }
    if (hi == 0) {  // lanes 0-15 hold rows M=0..7 (the real heads)
#pragma unroll
      for (int r = 0; r < 8; r++) simP[khalf][r * 64 + jt * 16 + lm] = acc[r];
    }
  }
  __syncthreads();

  // ---- masked softmax over j: wave h handles head h; lane covers j and j+32 ----
  {
    int h = wave;
    const int* mrow = mask + (size_t)n * JDIM;
    float v1 = simP[0][h * 64 + lane]      + simP[1][h * 64 + lane];
    float v2 = simP[0][h * 64 + 32 + lane] + simP[1][h * 64 + 32 + lane];
    if (mrow[lane] == 0)      v1 = -3.4028235e38f;
    if (mrow[32 + lane] == 0) v2 = -3.4028235e38f;
    float mx = fmaxf(v1, v2);
#pragma unroll
    for (int off = 16; off; off >>= 1) mx = fmaxf(mx, __shfl_xor(mx, off));
    float p1 = __expf(v1 - mx), p2 = __expf(v2 - mx);
    float s = p1 + p2;
#pragma unroll
    for (int off = 16; off; off >>= 1) s += __shfl_xor(s, off);
    float inv = 1.0f / s;
    attnS[h * 64 + lane]      = f2bf(p1 * inv);
    attnS[h * 64 + 32 + lane] = f2bf(p2 * inv);
  }
  __syncthreads();

  // ---- c GEMM: [16(8 used) x 512] = attn[16 x 64] x mems[64 x 512]; 8 waves x 4 n-tiles ----
  // B fragments now come from the transposed tile memsT (vector b64 loads, no gathers)
  unsigned short* Cn = Cbf + (size_t)n * PW;
#pragma unroll
  for (int q = 0; q < 4; q++) {
    int nt = wave + q * 8;
    v8f acc = (v8f){0.f,0.f,0.f,0.f,0.f,0.f,0.f,0.f};
#pragma unroll
    for (int kk = 0; kk < 2; kk++) {
      int k0 = kk * 32;
      const unsigned short* ar = &attnS[lm * 64 + k0 + (hi ? 8 : 0)];
      bf16x16 af = bfpair(*(const U128*)ar, *(const U128*)(ar + 16));
      int e  = nt * 16 + lm;
      int jb = k0 + (hi ? 8 : 0);
      const unsigned short* bp = &memsT[e * MTP + jb];
      union { U64 q4[4]; bf16x16 v; } bu;
      bu.q4[0] = *(const U64*)(bp);
      bu.q4[1] = *(const U64*)(bp + 4);
      bu.q4[2] = *(const U64*)(bp + 16);
      bu.q4[3] = *(const U64*)(bp + 20);
      acc = __builtin_amdgcn_wmma_f32_16x16x32_bf16(
          false, af, false, bu.v, (short)0, acc, false, false);
    }
    if (hi == 0) {
#pragma unroll
      for (int r = 0; r < 8; r++)
        Cn[r * 512 + nt * 16 + lm] = f2bf(acc[r]);
    }
  }
}

// ---------------- launch ----------------
extern "C" void kernel_launch(void* const* d_in, const int* in_sizes, int n_in,
                              void* d_out, int out_size, void* d_ws, size_t ws_size,
                              hipStream_t stream) {
  const float* x    = (const float*)d_in[0];  // [2048, 512]
  const float* mems = (const float*)d_in[1];  // [2048, 64, 512]
  const int*   mask = (const int*)d_in[2];    // [2048, 64]
  const float* Wq   = (const float*)d_in[3];  // [512, 512]
  const float* Wkv  = (const float*)d_in[4];  // [512, 1024]
  const float* Wo   = (const float*)d_in[5];  // [512, 512]
  const float* bo   = (const float*)d_in[6];  // [512]
  float* out = (float*)d_out;                 // [2048, 512]

  // workspace layout (bf16 tensors), total 42 MB
  char* ws = (char*)d_ws;
  unsigned short* Xbf  = (unsigned short*)(ws);                          //  2 MB [2048,512]
  unsigned short* Abf  = (unsigned short*)(ws + (size_t)(2)  * 1048576); //  4 MB [512,4096]
  unsigned short* W2bf = (unsigned short*)(ws + (size_t)(6)  * 1048576); //  4 MB [4096,512]
  unsigned short* Pbf  = (unsigned short*)(ws + (size_t)(10) * 1048576); // 16 MB [2048,4096]
  unsigned short* Cbf  = (unsigned short*)(ws + (size_t)(26) * 1048576); // 16 MB [2048,4096]

  f32_to_bf16_vec4<<<1024, 256, 0, stream>>>(x, Xbf, (NQ * DMODEL) / 4);
  prep_A <<<512, 256, 0, stream>>>(Wq, Wkv, Abf);
  prep_W2<<<512, 256, 0, stream>>>(Wkv, Wo, W2bf);
  // P = X @ Astack   [2048 x 4096], K=512
  gemm_bf16<true ><<<dim3(PW / 128, NQ / 128), 256, 0, stream>>>(
      Xbf, Abf, Pbf, nullptr, NQ, PW, DMODEL);
  // fused attention per query row (streams mems exactly once)
  attn_kernel<<<NQ, 256, 0, stream>>>(mems, mask, Pbf, Cbf);
  // out = C @ W2stack + bo   [2048 x 512], K=4096
  gemm_bf16<false><<<dim3(DMODEL / 128, NQ / 128), 256, 0, stream>>>(
      Cbf, W2bf, out, bo, NQ, DMODEL, PW);
}
